// Seq_Model_29343216566492
// MI455X (gfx1250) — matrix operand order repaired and checked by reading, not compiled, not run on hardware
//
#include <hip/hip_runtime.h>
#include <hip/hip_bf16.h>

typedef __attribute__((ext_vector_type(16))) _Float16 v16h;
typedef __attribute__((ext_vector_type(8)))  float    v8f;

#define B_   64
#define T_   2048
#define I_   150
#define BT_  (B_ * T_)
#define NG   128          // 64 fwd gates + 64 bwd gates
#define KPAD 160          // 5 K-chunks of 32
#define KSTR 164          // LDS row stride in halves (bank-conflict pad)

__device__ __forceinline__ float sigmoidf(float x) {
    return 1.0f / (1.0f + __expf(-x));
}

// ---------------------------------------------------------------------------
// Kernel 1: fused bidirectional layer-1 input projection via WMMA.
// gates[bt][0..63]  = x[bt] @ w_ih1_f^T + b1_f   (order i,f,g,o x16)
// gates[bt][64..127]= x[bt] @ w_ih1_b^T + b1_b
// Workgroup = 256 threads (8 waves). Each wave owns a 16(M)x128(N) strip.
// ---------------------------------------------------------------------------
__global__ __launch_bounds__(256)
void gates1_wmma(const float* __restrict__ x,
                 const float* __restrict__ w_f, const float* __restrict__ b_f,
                 const float* __restrict__ w_b, const float* __restrict__ b_b,
                 float* __restrict__ gates)
{
    __shared__ _Float16 wl[NG * KSTR];   // ~41 KB: both dirs' weights as f16
    const int tid = threadIdx.x;

    for (int idx = tid; idx < NG * KPAD; idx += 256) {
        int n = idx / KPAD, k = idx - n * KPAD;
        float v = 0.0f;
        if (k < I_) v = (n < 64) ? w_f[n * I_ + k] : w_b[(n - 64) * I_ + k];
        wl[n * KSTR + k] = (_Float16)v;
    }
    __syncthreads();

    const int wave  = tid >> 5;
    const int lane  = tid & 31;
    const int lhalf = lane >> 4;       // which 16-lane half
    const int m     = lane & 15;
    const long Mbase = (long)blockIdx.x * 128 + wave * 16;

    v8f acc[8] = {};

    for (int kc = 0; kc < KPAD / 32; ++kc) {
        // A fragment (16x32 f16): row M = Mbase+m for both lane halves;
        // element e -> K = kc*32 + lhalf*16 + e
        v16h a;
        const float* xrow = x + (Mbase + m) * (long)I_;
        #pragma unroll
        for (int e = 0; e < 16; ++e) {
            int k = kc * 32 + (lhalf << 4) + e;
            a[e] = (k < I_) ? (_Float16)xrow[k] : (_Float16)0.0f;
        }
        #pragma unroll
        for (int nt = 0; nt < 8; ++nt) {
            // B fragment (32x16 f16): column N = nt*16+m; K = kc*32 + lhalf*16 + e
            v16h bf;
            const _Float16* wrow = &wl[(nt * 16 + m) * KSTR + kc * 32 + (lhalf << 4)];
            #pragma unroll
            for (int e = 0; e < 16; ++e) bf[e] = wrow[e];
            acc[nt] = __builtin_amdgcn_wmma_f32_16x16x32_f16(
                false, a, false, bf, (short)0, acc[nt], false, false);
        }
    }

    // C/D layout: lane half selects M+8; VGPR r = M row offset; col = lane&15
    #pragma unroll
    for (int nt = 0; nt < 8; ++nt) {
        int col = nt * 16 + m;
        float bias = (col < 64) ? b_f[col] : b_b[col - 64];
        #pragma unroll
        for (int r = 0; r < 8; ++r) {
            long row = Mbase + r + 8 * lhalf;
            gates[row * NG + col] = acc[nt][r] + bias;
        }
    }
}

// ---------------------------------------------------------------------------
// Kernel 2: layer-1 recurrence. One 1024-thread workgroup per direction.
// thread = (b, h); h-state in LDS; c-state in registers.
// ---------------------------------------------------------------------------
__global__ __launch_bounds__(1024)
void lstm1_rec(const float* __restrict__ gates,
               const float* __restrict__ w_hh_f,
               const float* __restrict__ w_hh_b,
               float* __restrict__ h1)
{
    const int dir = blockIdx.x;          // 0 = forward, 1 = backward
    const int tid = threadIdx.x;
    const int b = tid >> 4;
    const int h = tid & 15;

    __shared__ float whh[64 * 16];       // rows: i(0..15) f(16..31) g(32..47) o(48..63)
    __shared__ float hs[64][16];

    const float* w = dir ? w_hh_b : w_hh_f;
    for (int idx = tid; idx < 64 * 16; idx += 1024) whh[idx] = w[idx];
    hs[b][h] = 0.0f;
    float c = 0.0f;
    __syncthreads();

    for (int t = 0; t < T_; ++t) {
        const int time = dir ? (T_ - 1 - t) : t;
        const float* g4 = gates + ((long)b * T_ + time) * NG + dir * 64;

        float hv[16];
        #pragma unroll
        for (int k = 0; k < 16; ++k) hv[k] = hs[b][k];

        float gi = g4[h], gf = g4[16 + h], gg = g4[32 + h], go = g4[48 + h];
        #pragma unroll
        for (int k = 0; k < 16; ++k) {
            gi = __builtin_fmaf(whh[h * 16 + k],        hv[k], gi);
            gf = __builtin_fmaf(whh[(16 + h) * 16 + k], hv[k], gf);
            gg = __builtin_fmaf(whh[(32 + h) * 16 + k], hv[k], gg);
            go = __builtin_fmaf(whh[(48 + h) * 16 + k], hv[k], go);
        }
        c = sigmoidf(gf) * c + sigmoidf(gi) * tanhf(gg);
        float hn = sigmoidf(go) * tanhf(c);

        __syncthreads();                 // all reads of hs done
        hs[b][h] = hn;
        h1[((long)b * T_ + time) * 32 + dir * 16 + h] = hn;
        __syncthreads();                 // new hs visible
    }
}

// ---------------------------------------------------------------------------
// Kernel 3: layer-2 input projection: xg2[bt][dir*4+r] = h1[bt](32) . w2 + b2
// ---------------------------------------------------------------------------
__global__ __launch_bounds__(256)
void gates2(const float* __restrict__ h1,
            const float* __restrict__ w2f, const float* __restrict__ b2f,
            const float* __restrict__ w2b, const float* __restrict__ b2b,
            float* __restrict__ xg2)
{
    long idx = (long)blockIdx.x * 256 + threadIdx.x;   // over BT*8
    if (idx >= (long)BT_ * 8) return;
    long bt = idx >> 3;
    int col = (int)(idx & 7);
    int dir = col >> 2, r = col & 3;
    const float* w = (dir ? w2b : w2f) + r * 32;
    const float* hrow = h1 + bt * 32;
    float s = (dir ? b2b : b2f)[r];
    #pragma unroll
    for (int k = 0; k < 32; ++k) s = __builtin_fmaf(w[k], hrow[k], s);
    xg2[idx] = s;
}

// ---------------------------------------------------------------------------
// Kernel 4: layer-2 recurrence (H2=1). 128 independent scalar scans.
// ---------------------------------------------------------------------------
__global__ __launch_bounds__(128)
void lstm2_rec(const float* __restrict__ xg2,
               const float* __restrict__ whh2f,
               const float* __restrict__ whh2b,
               float* __restrict__ h2)
{
    const int tid = threadIdx.x;
    const int b = tid >> 1, dir = tid & 1;
    const float* wr = dir ? whh2b : whh2f;   // [4,1]
    const float w0 = wr[0], w1 = wr[1], w2 = wr[2], w3 = wr[3];
    float h = 0.0f, c = 0.0f;
    for (int t = 0; t < T_; ++t) {
        const int time = dir ? (T_ - 1 - t) : t;
        const float* g = xg2 + ((long)b * T_ + time) * 8 + dir * 4;
        float gi = __builtin_fmaf(h, w0, g[0]);
        float gf = __builtin_fmaf(h, w1, g[1]);
        float gg = __builtin_fmaf(h, w2, g[2]);
        float go = __builtin_fmaf(h, w3, g[3]);
        c = sigmoidf(gf) * c + sigmoidf(gi) * tanhf(gg);
        h = sigmoidf(go) * tanhf(c);
        h2[((long)b * T_ + time) * 2 + dir] = h;
    }
}

// ---------------------------------------------------------------------------
// Kernel 5: FC + softmax over the SEQUENCE axis. One workgroup per (b, class).
// ---------------------------------------------------------------------------
__global__ __launch_bounds__(256)
void fc_softmax(const float* __restrict__ h2,
                const float* __restrict__ fc_w, const float* __restrict__ fc_b,
                float* __restrict__ out)
{
    __shared__ float red[256];
    const int b = blockIdx.x / 3, c = blockIdx.x % 3;
    const int tid = threadIdx.x;
    const float w0 = fc_w[c * 2], w1 = fc_w[c * 2 + 1], bias = fc_b[c];
    const float* hb = h2 + (long)b * T_ * 2;

    float m = -INFINITY;
    for (int t = tid; t < T_; t += 256) {
        float l = __builtin_fmaf(hb[t * 2], w0, __builtin_fmaf(hb[t * 2 + 1], w1, bias));
        m = fmaxf(m, l);
    }
    red[tid] = m; __syncthreads();
    for (int s = 128; s > 0; s >>= 1) {
        if (tid < s) red[tid] = fmaxf(red[tid], red[tid + s]);
        __syncthreads();
    }
    m = red[0]; __syncthreads();

    float sum = 0.0f;
    for (int t = tid; t < T_; t += 256) {
        float l = __builtin_fmaf(hb[t * 2], w0, __builtin_fmaf(hb[t * 2 + 1], w1, bias));
        sum += __expf(l - m);
    }
    red[tid] = sum; __syncthreads();
    for (int s = 128; s > 0; s >>= 1) {
        if (tid < s) red[tid] += red[tid + s];
        __syncthreads();
    }
    const float inv = 1.0f / red[0];

    for (int t = tid; t < T_; t += 256) {
        float l = __builtin_fmaf(hb[t * 2], w0, __builtin_fmaf(hb[t * 2 + 1], w1, bias));
        out[((long)b * T_ + t) * 3 + c] = __expf(l - m) * inv;
    }
}

// ---------------------------------------------------------------------------
extern "C" void kernel_launch(void* const* d_in, const int* in_sizes, int n_in,
                              void* d_out, int out_size, void* d_ws, size_t ws_size,
                              hipStream_t stream) {
    (void)in_sizes; (void)n_in; (void)out_size; (void)ws_size;

    const float* x       = (const float*)d_in[0];
    const float* w_ih1_f = (const float*)d_in[1];
    const float* w_hh1_f = (const float*)d_in[2];
    const float* b1_f    = (const float*)d_in[3];
    const float* w_ih1_b = (const float*)d_in[4];
    const float* w_hh1_b = (const float*)d_in[5];
    const float* b1_b    = (const float*)d_in[6];
    const float* w_ih2_f = (const float*)d_in[7];
    const float* w_hh2_f = (const float*)d_in[8];
    const float* b2_f    = (const float*)d_in[9];
    const float* w_ih2_b = (const float*)d_in[10];
    const float* w_hh2_b = (const float*)d_in[11];
    const float* b2_b    = (const float*)d_in[12];
    const float* fc_w    = (const float*)d_in[13];
    const float* fc_b    = (const float*)d_in[14];
    float* out = (float*)d_out;

    char* ws = (char*)d_ws;
    float* gates = (float*)(ws);                                   // BT*128 f32
    float* h1b   = (float*)(ws + (size_t)BT_ * NG * 4);            // BT*32
    float* xg2b  = (float*)(ws + (size_t)BT_ * (NG + 32) * 4);     // BT*8
    float* h2b   = (float*)(ws + (size_t)BT_ * (NG + 40) * 4);     // BT*2

    gates1_wmma<<<dim3(BT_ / 128), dim3(256), 0, stream>>>(
        x, w_ih1_f, b1_f, w_ih1_b, b1_b, gates);

    lstm1_rec<<<dim3(2), dim3(1024), 0, stream>>>(gates, w_hh1_f, w_hh1_b, h1b);

    gates2<<<dim3((BT_ * 8) / 256), dim3(256), 0, stream>>>(
        h1b, w_ih2_f, b2_f, w_ih2_b, b2_b, xg2b);

    lstm2_rec<<<dim3(1), dim3(128), 0, stream>>>(xg2b, w_hh2_f, w_hh2_b, h2b);

    fc_softmax<<<dim3(B_ * 3), dim3(256), 0, stream>>>(h2b, fc_w, fc_b, out);
}
